// GATEncoder_29437705846981
// MI455X (gfx1250) — matrix operand order repaired and checked by reading, not compiled, not run on hardware
//
#include <hip/hip_runtime.h>
#include <math.h>

// Problem constants (from the reference)
#define N_NODES   10000
#define N_EDGES   100000
#define G_IN      2000
#define H_HEADS   20
#define C_CH      128
#define HC        2560                   // H * C (same for both layers)
#define EP        (N_EDGES + N_NODES)    // edges + self loops
#define NEG_SLOPE 0.2f
#define MPAD      10048                  // 157 * 64
#define KPAD1     2016                   // ceil(2000/32)*32

typedef __attribute__((ext_vector_type(16))) __bf16 bf16x16;
typedef __attribute__((ext_vector_type(8)))  __bf16 bf16x8;
typedef __attribute__((ext_vector_type(8)))  float  f32x8;
typedef __attribute__((ext_vector_type(4)))  unsigned int u32x4;

union FragU  { bf16x16 f; bf16x8 h[2]; };
union Pack8  { bf16x8 v; u32x4 u; };

// ---------------------------------------------------------------------------
// bf16 WMMA GEMM:  C[M,Nn] = A[M,:] * B^T[Nn,:]^T
//   A  : bf16 [MPAD][lda] row-major, K zero-padded to lda (multiple of 32)
//   BT : bf16 [Nn][lda]   (B stored transposed, K-contiguous), zero-padded
// Block: 256 thr = 8 wave32s; block tile 64(M) x 128(N); wave tile 32x32.
// No bounds checks in the K loop; all LDS traffic is b128.
// ---------------------------------------------------------------------------
__launch_bounds__(256)
__global__ void gemm_bf16_wmma(const __bf16* __restrict__ A,
                               const __bf16* __restrict__ BT,
                               float* __restrict__ C,
                               int M, int lda, int Nn) {
  __shared__ __bf16 As[64][40];     // 32 K + pad; 80B row stride (16B aligned)
  __shared__ __bf16 Bs[128][40];    // transposed tile: [n][k]

  const int tid  = threadIdx.x;
  const int lane = tid & 31;
  const int wave = tid >> 5;            // 0..7
  const int wm   = (wave >> 2) * 32;    // 0 or 32
  const int wn   = (wave & 3)  * 32;    // 0,32,64,96
  const int m0   = blockIdx.x * 64;
  const int n0   = blockIdx.y * 128;

  const int rsel = lane & 15;
  const int kbA  = (lane < 16) ? 0 : 8;    // ISA 16-bit A fragment K base
  const int kbB  = (lane < 16) ? 0 : 8;    // ISA 16-bit B fragment K base

  // staging coordinates (constant over K loop)
  const int ar = tid >> 2;              // 0..63
  const int ac = (tid & 3) * 8;         // 0,8,16,24
  const int bn = tid >> 1;              // 0..127
  const int bk = (tid & 1) * 16;        // 0,16

  const __bf16* Ap  = A  + (long long)(m0 + ar) * lda + ac;
  const __bf16* BTp = BT + (long long)(n0 + bn) * lda + bk;

  f32x8 acc[2][2] = {};

  for (int kk = 0; kk < lda; kk += 32) {
    // ---- stage tiles: pure b128 copies, no predication
    *(u32x4*)&As[ar][ac]     = *(const u32x4*)(Ap + kk);
    *(u32x4*)&Bs[bn][bk]     = *(const u32x4*)(BTp + kk);
    *(u32x4*)&Bs[bn][bk + 8] = *(const u32x4*)(BTp + kk + 8);
    if (kk + 32 < lda) {                 // prefetch next K tile
      __builtin_prefetch(Ap + kk + 32, 0, 1);
      __builtin_prefetch(BTp + kk + 32, 0, 1);
    }
    __syncthreads();

    // ---- fragments: each half is one 16-byte LDS load
    FragU a[2], b[2];
#pragma unroll
    for (int mi = 0; mi < 2; ++mi) {
      int row = wm + mi * 16 + rsel;
      a[mi].h[0] = *(const bf16x8*)&As[row][kbA];        // K = base+0..7
      a[mi].h[1] = *(const bf16x8*)&As[row][16 + kbA];   // K = base+16..23
    }
#pragma unroll
    for (int ni = 0; ni < 2; ++ni) {
      int col = wn + ni * 16 + rsel;
      b[ni].h[0] = *(const bf16x8*)&Bs[col][kbB];        // K = base+0..7
      b[ni].h[1] = *(const bf16x8*)&Bs[col][16 + kbB];   // K = base+16..23
    }

#pragma unroll
    for (int mi = 0; mi < 2; ++mi)
#pragma unroll
      for (int ni = 0; ni < 2; ++ni)
        acc[mi][ni] = __builtin_amdgcn_wmma_f32_16x16x32_bf16(
            false, a[mi].f, false, b[ni].f,
            (short)0, acc[mi][ni], false, false);

    __syncthreads();
  }

  // ---- store D (16x16 f32 C/D layout: lane%16 = N col; lanes>=16 -> M+8)
#pragma unroll
  for (int mi = 0; mi < 2; ++mi)
#pragma unroll
    for (int ni = 0; ni < 2; ++ni) {
      int col   = n0 + wn + ni * 16 + rsel;
      int rbase = m0 + wm + mi * 16 + ((lane < 16) ? 0 : 8);
#pragma unroll
      for (int v = 0; v < 8; ++v) {
        int row = rbase + v;
        if (row < M) C[(long long)row * Nn + col] = acc[mi][ni][v];
      }
    }
}

// ---------------------------------------------------------------------------
// fp32 [M][K] -> bf16 [MPAD][Kpad] row-major, zero-padded. 8 elems/thread.
// K must be a multiple of 8 (2000 and 2560 both are) -> fast vector path.
// ---------------------------------------------------------------------------
__launch_bounds__(256)
__global__ void conv_a_bf16(const float* __restrict__ A, __bf16* __restrict__ out,
                            int M, int K, int Kpad) {
  int t = blockIdx.x * blockDim.x + threadIdx.x;
  int kg = Kpad >> 3;
  if (t >= MPAD * kg) return;
  int r  = t / kg;
  int c0 = (t % kg) * 8;
  Pack8 o;
  if (r < M && c0 + 8 <= K) {
    const float4* p = (const float4*)(A + (long long)r * K + c0);
    float4 f0 = p[0], f1 = p[1];
    o.v[0] = (__bf16)f0.x; o.v[1] = (__bf16)f0.y;
    o.v[2] = (__bf16)f0.z; o.v[3] = (__bf16)f0.w;
    o.v[4] = (__bf16)f1.x; o.v[5] = (__bf16)f1.y;
    o.v[6] = (__bf16)f1.z; o.v[7] = (__bf16)f1.w;
  } else {
#pragma unroll
    for (int j = 0; j < 8; ++j) o.v[j] = (__bf16)0.0f;
  }
  *(u32x4*)(out + (long long)r * Kpad + c0) = o.u;
}

// W fp32 [K][Nn] -> BT bf16 [Nn][Kpad] (transposed), zero-padded K.
// Lanes own consecutive n -> each of the 8 reads is coalesced.
__launch_bounds__(256)
__global__ void conv_bt_bf16(const float* __restrict__ W, __bf16* __restrict__ BT,
                             int K, int Kpad, int Nn) {
  int t = blockIdx.x * blockDim.x + threadIdx.x;
  int kg = Kpad >> 3;
  if (t >= Nn * kg) return;
  int n  = t % Nn;
  int k0 = (t / Nn) * 8;
  Pack8 o;
#pragma unroll
  for (int j = 0; j < 8; ++j) {
    int k = k0 + j;
    float f = (k < K) ? W[(long long)k * Nn + n] : 0.0f;
    o.v[j] = (__bf16)f;
  }
  *(u32x4*)(BT + (long long)n * Kpad + k0) = o.u;
}

// agg fp32 [N,HC] --(+bias, relu)--> bf16 [MPAD][HC] (layer-2 GEMM input)
__launch_bounds__(256)
__global__ void bias_relu_bf16_kernel(const float* __restrict__ agg,
                                      const float* __restrict__ bias,
                                      __bf16* __restrict__ out) {
  int t = blockIdx.x * blockDim.x + threadIdx.x;
  const int kg = HC >> 3;
  if (t >= MPAD * kg) return;
  int r  = t / kg;
  int c0 = (t % kg) * 8;
  Pack8 o;
  if (r < N_NODES) {
    const float* p = agg + (long long)r * HC + c0;
    const float* bp = bias + c0;
#pragma unroll
    for (int j = 0; j < 8; ++j) {
      float v = p[j] + bp[j];
      o.v[j] = (__bf16)fmaxf(v, 0.0f);
    }
  } else {
#pragma unroll
    for (int j = 0; j < 8; ++j) o.v[j] = (__bf16)0.0f;
  }
  *(u32x4*)(out + (long long)r * HC + c0) = o.u;
}

// ---------------------------------------------------------------------------
// Per-(node,head) attention dots, one wave32 each; also re-init emax/denom.
// ---------------------------------------------------------------------------
__launch_bounds__(256)
__global__ void node_att_kernel(const float* __restrict__ hbuf,
                                const float* __restrict__ att_s,
                                const float* __restrict__ att_d,
                                float* __restrict__ a_src,
                                float* __restrict__ a_dst,
                                float* __restrict__ emax,
                                float* __restrict__ denom) {
  int gw   = (blockIdx.x * blockDim.x + threadIdx.x) >> 5;
  int lane = threadIdx.x & 31;
  if (gw >= N_NODES * H_HEADS) return;
  int n  = gw / H_HEADS;
  int hd = gw % H_HEADS;
  const float* hp  = hbuf + (long long)n * HC + hd * C_CH;
  const float* asp = att_s + hd * C_CH;
  const float* adp = att_d + hd * C_CH;
  float ss = 0.f, sd = 0.f;
#pragma unroll
  for (int j = 0; j < 4; ++j) {
    int c    = lane + 32 * j;
    float hv = hp[c];
    ss += hv * asp[c];
    sd += hv * adp[c];
  }
#pragma unroll
  for (int off = 16; off > 0; off >>= 1) {
    ss += __shfl_xor(ss, off, 32);
    sd += __shfl_xor(sd, off, 32);
  }
  if (lane == 0) {
    a_src[gw] = ss;
    a_dst[gw] = sd;
    emax[gw]  = -__builtin_inff();
    denom[gw] = 0.f;
  }
}

__device__ __forceinline__ void edge_sd(const int* __restrict__ ei, int e,
                                        int& s, int& d) {
  if (e < N_EDGES) { s = ei[e]; d = ei[N_EDGES + e]; }
  else             { s = e - N_EDGES; d = s; }       // self loops
}

// e = leaky_relu(a_src[src] + a_dst[dst]); segment-max via int-punned atomics
__launch_bounds__(256)
__global__ void edge_logit_kernel(const int* __restrict__ ei,
                                  const float* __restrict__ a_src,
                                  const float* __restrict__ a_dst,
                                  float* __restrict__ buf_e,
                                  float* __restrict__ emax) {
  int t = blockIdx.x * blockDim.x + threadIdx.x;
  if (t >= EP * H_HEADS) return;
  int e  = t / H_HEADS;
  int hd = t % H_HEADS;
  int s, d; edge_sd(ei, e, s, d);
  float v = a_src[s * H_HEADS + hd] + a_dst[d * H_HEADS + hd];
  v = (v > 0.f) ? v : NEG_SLOPE * v;
  buf_e[t] = v;
  float* addr = &emax[d * H_HEADS + hd];
  if (v >= 0.f) atomicMax((int*)addr, __float_as_int(v));
  else          atomicMin((unsigned int*)addr, __float_as_uint(v));
}

// ex = exp(e - emax[dst]); denom[dst] += ex
__launch_bounds__(256)
__global__ void edge_exp_kernel(const int* __restrict__ ei,
                                const float* __restrict__ emax,
                                float* __restrict__ buf_e,
                                float* __restrict__ denom) {
  int t = blockIdx.x * blockDim.x + threadIdx.x;
  if (t >= EP * H_HEADS) return;
  int e  = t / H_HEADS;
  int hd = t % H_HEADS;
  int s, d; edge_sd(ei, e, s, d);
  (void)s;
  float ex = __expf(buf_e[t] - emax[d * H_HEADS + hd]);
  buf_e[t] = ex;
  atomicAdd(&denom[d * H_HEADS + hd], ex);
}

__launch_bounds__(256)
__global__ void zero_kernel(float* __restrict__ p, int n) {
  int i = blockIdx.x * blockDim.x + threadIdx.x;
  if (i < n) p[i] = 0.f;
}

// Scatter: agg[dst,h,:] += alpha * h[src,h,:]; wave32 per (edge,head).
// Target (102MB) fits in the 192MB global L2 -> atomics stay on-chip.
__launch_bounds__(256)
__global__ void aggregate_kernel(const int* __restrict__ ei,
                                 const float* __restrict__ buf_e,
                                 const float* __restrict__ denom,
                                 const float* __restrict__ hbuf,
                                 float* __restrict__ agg) {
  int gw   = (blockIdx.x * blockDim.x + threadIdx.x) >> 5;
  int lane = threadIdx.x & 31;
  if (gw >= EP * H_HEADS) return;
  int e  = gw / H_HEADS;
  int hd = gw % H_HEADS;
  int s, d; edge_sd(ei, e, s, d);
  float alpha = buf_e[gw] / denom[d * H_HEADS + hd];
  const float* hp = hbuf + (long long)s * HC + hd * C_CH;
  float*       op = agg  + (long long)d * HC + hd * C_CH;
#pragma unroll
  for (int j = 0; j < 4; ++j) {
    int c = lane + 32 * j;
    atomicAdd(&op[c], alpha * hp[c]);
  }
}

// out[n,c] = relu(mean_h(agg[n,h,c]) + b2[c])
__launch_bounds__(256)
__global__ void head_mean_kernel(const float* __restrict__ agg,
                                 const float* __restrict__ b2,
                                 float* __restrict__ out) {
  int t = blockIdx.x * blockDim.x + threadIdx.x;
  if (t >= N_NODES * C_CH) return;
  int n = t / C_CH;
  int c = t % C_CH;
  float s = 0.f;
#pragma unroll
  for (int hd = 0; hd < H_HEADS; ++hd)
    s += agg[(long long)n * HC + hd * C_CH + c];
  float v = s * (1.0f / H_HEADS) + b2[c];
  out[t] = (v > 0.f) ? v : 0.f;
}

// ---------------------------------------------------------------------------
extern "C" void kernel_launch(void* const* d_in, const int* in_sizes, int n_in,
                              void* d_out, int out_size, void* d_ws, size_t ws_size,
                              hipStream_t stream) {
  const float* x        = (const float*)d_in[0];
  const int*   eidx     = (const int*)  d_in[1];
  const float* W1       = (const float*)d_in[2];
  const float* att_src1 = (const float*)d_in[3];
  const float* att_dst1 = (const float*)d_in[4];
  const float* b1       = (const float*)d_in[5];
  const float* W2       = (const float*)d_in[6];
  const float* att_src2 = (const float*)d_in[7];
  const float* att_dst2 = (const float*)d_in[8];
  const float* b2       = (const float*)d_in[9];
  (void)in_sizes; (void)n_in; (void)out_size; (void)ws_size;

  // fp32 workspace region (~217MB), then bf16 region (~75MB); all 16B aligned
  float* ws    = (float*)d_ws;
  float* buf_h = ws;                                   // [N, HC] features h
  float* buf_x = buf_h + (long long)N_NODES * HC;      // [N, HC] agg target
  float* a_src = buf_x + (long long)N_NODES * HC;      // [N, H]
  float* a_dst = a_src + N_NODES * H_HEADS;
  float* emax  = a_dst + N_NODES * H_HEADS;
  float* denom = emax  + N_NODES * H_HEADS;
  float* buf_e = denom + N_NODES * H_HEADS;            // [EP, H]
  __bf16* Abf  = (__bf16*)(buf_e + (long long)EP * H_HEADS);  // [MPAD][<=HC]
  __bf16* BT1  = Abf + (long long)MPAD * HC;           // [HC][KPAD1]
  __bf16* BT2  = BT1 + (long long)HC * KPAD1;          // [HC][HC]

  const dim3 blk(256);
  const dim3 ggemm(MPAD / 64, HC / 128);
  const int nh_waves = N_NODES * H_HEADS;
  const int eh       = EP * H_HEADS;
  const int nhc      = N_NODES * HC;
  const int convA1_n = MPAD * (KPAD1 / 8);
  const int convB1_n = HC * (KPAD1 / 8);
  const int convB2_n = HC * (HC / 8);
  const int convA2_n = MPAD * (HC / 8);

  // ---- one-shot operand conversion (bf16 + padding + W transpose)
  conv_a_bf16 <<<(convA1_n + 255) / 256, blk, 0, stream>>>(x,  Abf, N_NODES, G_IN, KPAD1);
  conv_bt_bf16<<<(convB1_n + 255) / 256, blk, 0, stream>>>(W1, BT1, G_IN, KPAD1, HC);
  conv_bt_bf16<<<(convB2_n + 255) / 256, blk, 0, stream>>>(W2, BT2, HC,   HC,    HC);

  // ---------------- layer 1 ----------------
  gemm_bf16_wmma<<<ggemm, blk, 0, stream>>>(Abf, BT1, buf_h, N_NODES, KPAD1, HC);
  node_att_kernel<<<(nh_waves * 32 + 255) / 256, blk, 0, stream>>>(
      buf_h, att_src1, att_dst1, a_src, a_dst, emax, denom);
  edge_logit_kernel<<<(eh + 255) / 256, blk, 0, stream>>>(eidx, a_src, a_dst, buf_e, emax);
  edge_exp_kernel  <<<(eh + 255) / 256, blk, 0, stream>>>(eidx, emax, buf_e, denom);
  zero_kernel      <<<(nhc + 255) / 256, blk, 0, stream>>>(buf_x, nhc);
  aggregate_kernel <<<(eh * 32 + 255) / 256, blk, 0, stream>>>(eidx, buf_e, denom, buf_h, buf_x);
  bias_relu_bf16_kernel<<<(convA2_n + 255) / 256, blk, 0, stream>>>(buf_x, b1, Abf);

  // ---------------- layer 2 ----------------
  gemm_bf16_wmma<<<ggemm, blk, 0, stream>>>(Abf, BT2, buf_h, N_NODES, HC, HC);
  node_att_kernel<<<(nh_waves * 32 + 255) / 256, blk, 0, stream>>>(
      buf_h, att_src2, att_dst2, a_src, a_dst, emax, denom);
  edge_logit_kernel<<<(eh + 255) / 256, blk, 0, stream>>>(eidx, a_src, a_dst, buf_e, emax);
  edge_exp_kernel  <<<(eh + 255) / 256, blk, 0, stream>>>(eidx, emax, buf_e, denom);
  zero_kernel      <<<(nhc + 255) / 256, blk, 0, stream>>>(buf_x, nhc);
  aggregate_kernel <<<(eh * 32 + 255) / 256, blk, 0, stream>>>(eidx, buf_e, denom, buf_h, buf_x);
  head_mean_kernel <<<(N_NODES * C_CH + 255) / 256, blk, 0, stream>>>(buf_x, b2, (float*)d_out);
}
